// LambdaLayer_618475291079
// MI455X (gfx1250) — matrix operand order repaired and checked by reading, not compiled, not run on hardware
//
#include <hip/hip_runtime.h>

// ---------------------------------------------------------------------------
// LambdaLayer on gfx1250.
// Heavy op: lam_p[b,n,k,v] = sum_m rel_emb[m-n+1023, k] * v[b,v,m]
//   -> per (b,n): 16x64 GEMM, K=1024, via V_WMMA_F32_16X16X32_BF16.
// rel_idx is Toeplitz (m-n+1023) so it is never read.
// ---------------------------------------------------------------------------

typedef __attribute__((ext_vector_type(8)))  float  v8f;
typedef __attribute__((ext_vector_type(16))) __bf16 v16bf;

#define NCTX 1024   // N = H*W
#define CCH  256    // C
#define NB   32     // B

__device__ __forceinline__ unsigned short f2bf(float f) {
  // round-to-nearest-even fp32 -> bf16
  unsigned int u = __float_as_uint(f);
  u += 0x7FFFu + ((u >> 16) & 1u);
  return (unsigned short)(u >> 16);
}

// ---------------------------------------------------------------------------
// Kernel 1: q/k/v 1x1-conv projections (+BN affine), also emit bf16 copy of v.
// grid (4, 9, 32), block 256. blockIdx.y selects a group of 16 output rows
// out of the 144 rows (64 q | 16 k | 64 v).
// ---------------------------------------------------------------------------
__global__ __launch_bounds__(256) void proj_kernel(
    const float* __restrict__ x,  const float* __restrict__ Wq,
    const float* __restrict__ Wk, const float* __restrict__ Wv,
    const float* __restrict__ gq, const float* __restrict__ bq,
    const float* __restrict__ gv, const float* __restrict__ bv,
    float* __restrict__ qbuf, float* __restrict__ kbuf,
    float* __restrict__ vbuf, unsigned short* __restrict__ vbf) {
  const int n  = blockIdx.x * 256 + threadIdx.x;
  const int og = blockIdx.y;                 // 0..8
  const int b  = blockIdx.z;
  const float* xb = x + (size_t)b * CCH * NCTX + n;

  const float* Wbase; int rowoff;
  if (og < 4)       { Wbase = Wq; rowoff = og * 16; }
  else if (og == 4) { Wbase = Wk; rowoff = 0; }
  else              { Wbase = Wv; rowoff = (og - 5) * 16; }

  float acc[16];
#pragma unroll
  for (int j = 0; j < 16; ++j) acc[j] = 0.f;

  for (int c = 0; c < CCH; ++c) {
    const float xv = xb[(size_t)c * NCTX];     // coalesced across n
#pragma unroll
    for (int j = 0; j < 16; ++j)
      acc[j] += Wbase[(rowoff + j) * CCH + c] * xv;  // uniform -> scalar loads
  }

  if (og < 4) {
#pragma unroll
    for (int j = 0; j < 16; ++j) {
      const int o = og * 16 + j;
      qbuf[((size_t)b * 64 + o) * NCTX + n] = acc[j] * gq[o] + bq[o];
    }
  } else if (og == 4) {
#pragma unroll
    for (int j = 0; j < 16; ++j)
      kbuf[((size_t)b * 16 + j) * NCTX + n] = acc[j];
  } else {
#pragma unroll
    for (int j = 0; j < 16; ++j) {
      const int vr = (og - 5) * 16 + j;
      const float val = acc[j] * gv[vr] + bv[vr];
      vbuf[((size_t)b * 64 + vr) * NCTX + n] = val;
      vbf [((size_t)b * 64 + vr) * NCTX + n] = f2bf(val);
    }
  }
}

// ---------------------------------------------------------------------------
// Kernel 2: softmax over the context axis (rows of 1024), in place on kbuf.
// grid (512), block 256.
// ---------------------------------------------------------------------------
__global__ __launch_bounds__(256) void softmax_kernel(float* __restrict__ kbuf) {
  float* r = kbuf + (size_t)blockIdx.x * NCTX;
  __shared__ float red[256];
  const int tid = threadIdx.x;

  float m = -3.4e38f;
  for (int i = tid; i < NCTX; i += 256) m = fmaxf(m, r[i]);
  red[tid] = m; __syncthreads();
  for (int s = 128; s > 0; s >>= 1) {
    if (tid < s) red[tid] = fmaxf(red[tid], red[tid + s]);
    __syncthreads();
  }
  const float M = red[0]; __syncthreads();

  float sum = 0.f;
  for (int i = tid; i < NCTX; i += 256) sum += __expf(r[i] - M);
  red[tid] = sum; __syncthreads();
  for (int s = 128; s > 0; s >>= 1) {
    if (tid < s) red[tid] += red[tid + s];
    __syncthreads();
  }
  const float inv = 1.0f / red[0];
  for (int i = tid; i < NCTX; i += 256) r[i] = __expf(r[i] - M) * inv;
}

// ---------------------------------------------------------------------------
// Kernel 3: lam_c[b,k,v] = sum_m sk[b,k,m] * v[b,v,m]   (fp32, tiny)
// grid (128), block 256: one thread per output element.
// ---------------------------------------------------------------------------
__global__ __launch_bounds__(256) void lamc_kernel(
    const float* __restrict__ sk, const float* __restrict__ vbuf,
    float* __restrict__ lamc) {
  const int id = blockIdx.x * 256 + threadIdx.x;  // b*1024 + k*64 + v
  const int b  = id >> 10;
  const int kk = (id >> 6) & 15;
  const int vv = id & 63;
  const float* srow = sk   + ((size_t)b * 16 + kk) * NCTX;
  const float* vrow = vbuf + ((size_t)b * 64 + vv) * NCTX;
  float s = 0.f;
  for (int m = 0; m < NCTX; ++m) s += srow[m] * vrow[m];
  lamc[id] = s;
}

// ---------------------------------------------------------------------------
// Kernel 4 (heavy): fused lam_p (WMMA bf16) + lam_c accumulate + q-contraction.
// grid (16, 32), block 256 = 8 waves. Workgroup = (batch b, 64 consecutive n).
// Wave w owns the stride-8 set n = nbase + w + 8*{0..7} and processes FOUR n
// per pass (distance 8 apart): 16 independent WMMA chains per m0-chunk share
// 4 B fragments, and the 4 A windows decompose into disjoint/shared 16B runs
// -> the scheduler fills all WMMA hazard windows with other WMMAs (no nops).
// LDS: rel_emb^T bf16 [16][2048] (64KB) + v_b bf16 [64][1024] (128KB).
// Both inner loops are kept rolled: full unroll makes the B ds_loads
// n-invariant and LICM spills 4KB/lane to scratch (seen in round 1).
// ---------------------------------------------------------------------------
__global__ __launch_bounds__(256, 1) void lambda_main_kernel(
    const float* __restrict__ rel_emb, const unsigned short* __restrict__ vbf,
    const float* __restrict__ qbuf, const float* __restrict__ lamc,
    float* __restrict__ out) {
  __shared__ unsigned short lds_re[16 * 2048];   // [k][idx], idx = m-n+1023
  __shared__ unsigned short lds_v [64 * 1024];   // [v][m]

  const int b   = blockIdx.y;
  const int tid = threadIdx.x;

  // Stage rel_emb (fp32 [2047][16]) transposed to bf16 [16][2048].
  for (int row = tid; row < 2 * NCTX - 1; row += 256) {
    const float* rr = rel_emb + row * 16;
#pragma unroll
    for (int kk = 0; kk < 16; ++kk)
      lds_re[kk * 2048 + row] = f2bf(rr[kk]);
  }
  // Stage this batch's v (bf16 [64][1024]) with 16B copies.
  {
    const unsigned short* src = vbf + (size_t)b * 64 * NCTX;
    for (int e = tid * 8; e < 64 * NCTX; e += 256 * 8)
      *(uint4*)&lds_v[e] = *(const uint4*)&src[e];
  }
  __syncthreads();

  const int lane = tid & 31;
  const int wave = tid >> 5;
  const int half = lane >> 4;   // K-half of the wave (contiguous k order used here)
  const int col  = lane & 15;   // M row for A, N column for B/C/D

  const unsigned short* rerow = &lds_re[col * 2048];

#pragma clang loop unroll(disable)
  for (int i = 0; i < 8; i += 4) {
    // This pass: n, n+8, n+16, n+24 (shared B, near-disjoint A windows).
    const int n = blockIdx.x * 64 + wave + 8 * i;

    // lam_c is n-invariant: it is the initial accumulator for every chain.
    // C/D layout: VGPR r, lane l -> row M = r + 8*(l>=16), col N = l%16.
    v8f acc[4][4];
#pragma unroll
    for (int t = 0; t < 4; ++t)
#pragma unroll
      for (int r = 0; r < 8; ++r) {
        const float c =
            lamc[((size_t)b * 16 + (r + 8 * half)) * 64 + t * 16 + col];
#pragma unroll
        for (int u = 0; u < 4; ++u) acc[u][t][r] = c;
      }

    const int idx0 = (NCTX - 1) - n;   // m=0 maps to rel row 1023-n (chain 0)

#pragma clang loop unroll(disable)
    for (int m0 = 0; m0 < NCTX; m0 += 32) {
      // A fragments: E_n^T [k=16 x m=32]; A[M=k][K=mc] = re_t[k][idx+m0+mc].
      // A k order used here: lane<16 -> K 0..15 ; lane>=16 -> K 16..31.
      // Chain u (n+8u) shifts the window by -8u: 16B runs at base-24..base+16.
      union { v16bf v; unsigned short s[16]; } A[4];
      const int base = idx0 + m0 + half * 16;
#pragma unroll
      for (int u = 0; u < 4; ++u) {
#pragma unroll
        for (int j = 0; j < 8; ++j) A[u].s[j]     = rerow[base - 8 * u + j];
#pragma unroll
        for (int j = 0; j < 8; ++j) A[u].s[8 + j] = rerow[base - 8 * u + 8 + j];
      }

#pragma unroll
      for (int t = 0; t < 4; ++t) {
        // B fragment: v [m=32 x v=16]; lane<16 -> K 0..15, lane>=16 -> K 16..31,
        // column = v-tile base + col. 32B contiguous in LDS; shared by 4 chains.
        const v16bf Bm =
            *(const v16bf*)&lds_v[(t * 16 + col) * NCTX + m0 + half * 16];
#pragma unroll
        for (int u = 0; u < 4; ++u)
          acc[u][t] = __builtin_amdgcn_wmma_f32_16x16x32_bf16(
              false, A[u].v, false, Bm, (short)0, acc[u][t], false, false);
      }
    }

    // Epilogue: Y[b, h*64+v, n] = sum_k q[b,h*16+k,n] * acc[k,v]  (fp32)
#pragma unroll
    for (int u = 0; u < 4; ++u) {
      const int nn = n + 8 * u;
      float qv[4][8];
#pragma unroll
      for (int h = 0; h < 4; ++h)
#pragma unroll
        for (int r = 0; r < 8; ++r)
          qv[h][r] = qbuf[((size_t)b * 64 + h * 16 + half * 8 + r) * NCTX + nn];

#pragma unroll
      for (int t = 0; t < 4; ++t) {
#pragma unroll
        for (int h = 0; h < 4; ++h) {
          float p = 0.f;
#pragma unroll
          for (int r = 0; r < 8; ++r) p += qv[h][r] * acc[u][t][r];
          p += __shfl_xor(p, 16, 32);          // combine the two k-halves
          if (lane < 16)
            out[((size_t)b * 256 + h * 64 + t * 16 + col) * NCTX + nn] = p;
        }
      }
    }
  }
}

// ---------------------------------------------------------------------------
// Launch.  Inputs (setup_inputs order):
//  0:x 1:Wq 2:Wk 3:Wv 4:gq 5:bq 6:gv 7:bv 8:rel_emb 9:rel_idx(unused)
// ---------------------------------------------------------------------------
extern "C" void kernel_launch(void* const* d_in, const int* in_sizes, int n_in,
                              void* d_out, int out_size, void* d_ws, size_t ws_size,
                              hipStream_t stream) {
  const float* x       = (const float*)d_in[0];
  const float* Wq      = (const float*)d_in[1];
  const float* Wk      = (const float*)d_in[2];
  const float* Wv      = (const float*)d_in[3];
  const float* gq      = (const float*)d_in[4];
  const float* bq      = (const float*)d_in[5];
  const float* gv      = (const float*)d_in[6];
  const float* bv      = (const float*)d_in[7];
  const float* rel_emb = (const float*)d_in[8];
  float* out = (float*)d_out;

  char* ws = (char*)d_ws;
  float* qbuf = (float*)ws;  ws += (size_t)NB * 64 * NCTX * sizeof(float);   // 8 MB
  float* kbuf = (float*)ws;  ws += (size_t)NB * 16 * NCTX * sizeof(float);   // 2 MB
  float* vbuf = (float*)ws;  ws += (size_t)NB * 64 * NCTX * sizeof(float);   // 8 MB
  float* lamc = (float*)ws;  ws += (size_t)NB * 16 * 64   * sizeof(float);   // 128 KB
  unsigned short* vbf = (unsigned short*)ws;                                  // 4 MB

  proj_kernel<<<dim3(4, 9, NB), 256, 0, stream>>>(
      x, Wq, Wk, Wv, gq, bq, gv, bv, qbuf, kbuf, vbuf, vbf);
  softmax_kernel<<<dim3(NB * 16), 256, 0, stream>>>(kbuf);
  lamc_kernel<<<dim3(NB * 16 * 64 / 256), 256, 0, stream>>>(kbuf, vbuf, lamc);
  lambda_main_kernel<<<dim3(16, NB), 256, 0, stream>>>(
      rel_emb, vbf, qbuf, lamc, out);
}
